// SRU_34437047779788
// MI455X (gfx1250) — matrix operand order repaired and checked by reading, not compiled, not run on hardware
//
#include <hip/hip_runtime.h>
#include <hip/hip_bf16.h>
#include <math.h>

typedef __attribute__((ext_vector_type(16))) _Float16 v16h;
typedef __attribute__((ext_vector_type(8)))  float    v8f;

#define S_DIM 2048
#define B_DIM 16
#define D_DIM 1024
#define M_DIM (S_DIM * B_DIM)   /* 32768 rows for both GEMMs */
#define N1_DIM (3 * D_DIM)      /* 3072 */

// ---------------------------------------------------------------------------
// Conversion / layout kernels
// ---------------------------------------------------------------------------
__global__ void cvt_f32_to_f16(const float* __restrict__ in,
                               _Float16* __restrict__ out, size_t n) {
  size_t i = (size_t)blockIdx.x * blockDim.x + threadIdx.x;
  if (i < n) out[i] = (_Float16)in[i];
}

// W: (D, 3D) row-major  ->  WhT: (3D, D) row-major, WhT[n][k] = W[k][n]
// (makes every lane's WMMA B-fragment two contiguous 8-half runs along K)
__global__ void transpose_W_to_f16(const float* __restrict__ W,
                                   _Float16* __restrict__ WhT) {
  size_t o = (size_t)blockIdx.x * blockDim.x + threadIdx.x;
  if (o >= (size_t)D_DIM * N1_DIM) return;
  int k = (int)(o & (size_t)(D_DIM - 1));
  int n = (int)(o >> 10);
  WhT[o] = (_Float16)W[(size_t)k * N1_DIM + n];
}

// ---------------------------------------------------------------------------
// WMMA fragment loaders (layouts per CDNA5 ISA 7.12.2, wave32)
// ---------------------------------------------------------------------------
// A 16x32 f16 (MxK): lane L<16 holds row M=L with K in {kh..kh+7, 16+kh..16+kh+7}
// where kh = 0; lanes 16-31 hold the same rows with kh = 8.
static __device__ __forceinline__ v16h load_a_frag(const _Float16* __restrict__ base,
                                                   int lda, int m0, int k0, int lane) {
  int r  = lane & 15;
  int kh = (lane >> 4) << 3;  // 0 or 8
  const _Float16* p = base + (size_t)(m0 + r) * lda + (size_t)(k0 + kh);
  union { uint4 u[2]; v16h h; } un;
  un.u[0] = *(const uint4*)(p);       // K = kh .. kh+7
  un.u[1] = *(const uint4*)(p + 16);  // K = 16+kh .. 16+kh+7
  return un.h;
}

// B 32x16 f16 (KxN): lanes 0-15 hold col N=lane, K={0..7,16..23}; lanes 16-31 K={8..15,24..31}.
// Source is the (N x K) transposed weight, so each lane's K-halves form two contiguous 8-half runs.
static __device__ __forceinline__ v16h load_b_frag(const _Float16* __restrict__ base,
                                                   int ldb, int n0, int k0, int lane) {
  int c  = lane & 15;
  int kb = (lane >> 4) << 3;  // 0 or 8
  const _Float16* p = base + (size_t)(n0 + c) * ldb + (size_t)(k0 + kb);
  union { uint4 u[2]; v16h h; } un;
  un.u[0] = *(const uint4*)(p);      // K = kb .. kb+7
  un.u[1] = *(const uint4*)(p + 16); // K = 16+kb .. 16+kb+7
  return un.h;
}

// ---------------------------------------------------------------------------
// GEMM1: Uh[m][n] = sum_k Xh[m][k] * WhT[n][k]     (M=32768, N=3072, K=1024)
// Each wave owns a 32x64 macro-tile (2 M-tiles x 4 N-tiles, A reused).
// Uh is streamed out with non-temporal stores so Xh/WhT stay L2-resident.
// ---------------------------------------------------------------------------
__global__ void gemm1_wmma(const _Float16* __restrict__ Xh,
                           const _Float16* __restrict__ WhT,
                           _Float16* __restrict__ Uh) {
  const int K = D_DIM, N = N1_DIM;
  int lane = threadIdx.x & 31;
  int wave = blockIdx.x * (blockDim.x >> 5) + (threadIdx.x >> 5);
  const int ntiles = N / 64;  // 48
  int m0 = (wave / ntiles) * 32;
  int n0 = (wave % ntiles) * 64;
  if (m0 >= M_DIM) return;

  v8f acc[2][4] = {};
  for (int k0 = 0; k0 < K; k0 += 32) {
    v16h a0 = load_a_frag(Xh, K, m0,      k0, lane);
    v16h a1 = load_a_frag(Xh, K, m0 + 16, k0, lane);
#pragma unroll
    for (int j = 0; j < 4; ++j) {
      v16h b = load_b_frag(WhT, K, n0 + 16 * j, k0, lane);
      acc[0][j] = __builtin_amdgcn_wmma_f32_16x16x32_f16(
          false, a0, false, b, (short)0, acc[0][j], false, false);
      acc[1][j] = __builtin_amdgcn_wmma_f32_16x16x32_f16(
          false, a1, false, b, (short)0, acc[1][j], false, false);
    }
  }

  // C/D layout: lane n = lane&15, VGPR v -> row v (+8 for lanes 16-31)
  int n  = lane & 15;
  int mo = (lane >> 4) << 3;
#pragma unroll
  for (int i = 0; i < 2; ++i)
#pragma unroll
    for (int j = 0; j < 4; ++j)
#pragma unroll
      for (int vv = 0; vv < 8; ++vv) {
        int m  = m0 + 16 * i + mo + vv;
        int nn = n0 + 16 * j + n;
        __builtin_nontemporal_store((_Float16)acc[i][j][vv],
                                    Uh + (size_t)m * N + nn);
      }
}

// ---------------------------------------------------------------------------
// SRU scan: 16384 independent chains (one per (b,d)), fp32 recurrence.
// Uh/X are streamed (non-temporal, single use); Hh stays regular-temporal
// because GEMM2 re-reads it 16x right after and it fits in L2 (64 MB).
// ---------------------------------------------------------------------------
__global__ void sru_scan(const _Float16* __restrict__ Uh,
                         const float* __restrict__ X,
                         const float* __restrict__ bias,
                         const float* __restrict__ v,
                         _Float16* __restrict__ Hh) {
  int idx = blockIdx.x * blockDim.x + threadIdx.x;  // 0 .. 16383
  int b = idx >> 10;
  int d = idx & (D_DIM - 1);
  const float vf = v[d], vr = v[D_DIM + d];
  const float bf = bias[d], br = bias[D_DIM + d];
  float c = 0.0f;
  for (int s = 0; s < S_DIM; ++s) {
    size_t rb = (size_t)s * B_DIM + b;
    size_t u3 = rb * N1_DIM + d;
    size_t rd = rb * D_DIM + d;
    if (s + 8 < S_DIM) {  // stream ahead: dependent loop, hide HBM latency
      size_t pu = ((size_t)(s + 8) * B_DIM + b) * N1_DIM + d;
      __builtin_prefetch(Uh + pu, 0, 0);
      __builtin_prefetch(Uh + pu + D_DIM, 0, 0);
      __builtin_prefetch(Uh + pu + 2 * D_DIM, 0, 0);
      __builtin_prefetch(X + ((size_t)(s + 8) * B_DIM + b) * D_DIM + d, 0, 0);
    }
    float uc = (float)__builtin_nontemporal_load(Uh + u3);
    float uf = (float)__builtin_nontemporal_load(Uh + u3 + D_DIM);
    float ur = (float)__builtin_nontemporal_load(Uh + u3 + 2 * D_DIM);
    float x  = __builtin_nontemporal_load(X + rd);
    float f = 1.0f / (1.0f + __expf(-(uf + vf * c + bf)));
    c = f * c + (1.0f - f) * uc;
    float r = 1.0f / (1.0f + __expf(-(ur + vr * c + br)));
    float h = r * c + (1.0f - r) * x;
    Hh[rd] = (_Float16)h;
  }
}

// ---------------------------------------------------------------------------
// GEMM2: Out[m][n] = sum_k Hh[m][k] * Wl[n][k] + bl[n]   (Wl.T GEMM, fp32 out)
// Out is never re-read -> non-temporal stores.
// ---------------------------------------------------------------------------
__global__ void gemm2_wmma(const _Float16* __restrict__ Hh,
                           const _Float16* __restrict__ Wlh,
                           const float* __restrict__ bl,
                           float* __restrict__ Out) {
  const int K = D_DIM, N = D_DIM;
  int lane = threadIdx.x & 31;
  int wave = blockIdx.x * (blockDim.x >> 5) + (threadIdx.x >> 5);
  const int ntiles = N / 64;  // 16
  int m0 = (wave / ntiles) * 32;
  int n0 = (wave % ntiles) * 64;
  if (m0 >= M_DIM) return;

  v8f acc[2][4] = {};
  for (int k0 = 0; k0 < K; k0 += 32) {
    v16h a0 = load_a_frag(Hh, K, m0,      k0, lane);
    v16h a1 = load_a_frag(Hh, K, m0 + 16, k0, lane);
#pragma unroll
    for (int j = 0; j < 4; ++j) {
      v16h b = load_b_frag(Wlh, K, n0 + 16 * j, k0, lane);
      acc[0][j] = __builtin_amdgcn_wmma_f32_16x16x32_f16(
          false, a0, false, b, (short)0, acc[0][j], false, false);
      acc[1][j] = __builtin_amdgcn_wmma_f32_16x16x32_f16(
          false, a1, false, b, (short)0, acc[1][j], false, false);
    }
  }

  int n  = lane & 15;
  int mo = (lane >> 4) << 3;
#pragma unroll
  for (int i = 0; i < 2; ++i)
#pragma unroll
    for (int j = 0; j < 4; ++j)
#pragma unroll
      for (int vv = 0; vv < 8; ++vv) {
        int m  = m0 + 16 * i + mo + vv;
        int nn = n0 + 16 * j + n;
        __builtin_nontemporal_store(acc[i][j][vv] + bl[nn],
                                    Out + (size_t)m * N + nn);
      }
}

// ---------------------------------------------------------------------------
// Launch
// ---------------------------------------------------------------------------
extern "C" void kernel_launch(void* const* d_in, const int* in_sizes, int n_in,
                              void* d_out, int out_size, void* d_ws, size_t ws_size,
                              hipStream_t stream) {
  (void)in_sizes; (void)n_in; (void)out_size; (void)ws_size;
  const float* X    = (const float*)d_in[0];  // (S,B,D)
  const float* W    = (const float*)d_in[1];  // (D,3D)
  const float* bias = (const float*)d_in[2];  // (2D)
  const float* v    = (const float*)d_in[3];  // (2D)
  const float* Wl   = (const float*)d_in[4];  // (D,D)
  const float* bl   = (const float*)d_in[5];  // (D)
  float* out = (float*)d_out;                 // (S,B,D) fp32

  // Workspace layout (f16 intermediates): ~328 MB total
  char* ws = (char*)d_ws;
  size_t off = 0;
  _Float16* Xh  = (_Float16*)(ws + off); off += (size_t)M_DIM  * D_DIM  * 2;  // 64 MB
  _Float16* WhT = (_Float16*)(ws + off); off += (size_t)N1_DIM * D_DIM  * 2;  //  6 MB
  _Float16* Wlh = (_Float16*)(ws + off); off += (size_t)D_DIM  * D_DIM  * 2;  //  2 MB
  _Float16* Uh  = (_Float16*)(ws + off); off += (size_t)M_DIM  * N1_DIM * 2;  // 192 MB
  _Float16* Hh  = (_Float16*)(ws + off); off += (size_t)M_DIM  * D_DIM  * 2;  // 64 MB

  // 1) casts / transposes
  size_t nX = (size_t)M_DIM * D_DIM;
  cvt_f32_to_f16<<<(unsigned)((nX + 255) / 256), 256, 0, stream>>>(X, Xh, nX);
  size_t nW = (size_t)D_DIM * N1_DIM;
  transpose_W_to_f16<<<(unsigned)((nW + 255) / 256), 256, 0, stream>>>(W, WhT);
  size_t nWl = (size_t)D_DIM * D_DIM;
  cvt_f32_to_f16<<<(unsigned)((nWl + 255) / 256), 256, 0, stream>>>(Wl, Wlh, nWl);

  // 2) U = X @ W  (f16 WMMA, f32 accum, f16 NT store)
  int waves1 = (M_DIM / 32) * (N1_DIM / 64);  // 49152 waves
  gemm1_wmma<<<waves1 / 8, 256, 0, stream>>>(Xh, WhT, Uh);

  // 3) sequential scan, 16384 independent (b,d) chains
  sru_scan<<<(B_DIM * D_DIM) / 256, 256, 0, stream>>>(Uh, X, bias, v, Hh);

  // 4) Out = Hseq @ Wl.T + bl
  int waves2 = (M_DIM / 32) * (D_DIM / 64);   // 16384 waves
  gemm2_wmma<<<waves2 / 8, 256, 0, stream>>>(Hh, Wlh, bl, out);
}
